// Generator_71330816852286
// MI455X (gfx1250) — compile-verified, matrix operand order checked by reading, not run on hardware
//
#include <hip/hip_runtime.h>
#include <math.h>

// ---------------- problem constants ----------------
#define B_    64
#define T_    600
#define K_    10
#define U_    64
#define C_    54
#define H_    512
#define OUT_  121
// unified activation layout: [x(0:3) | w(3:57) | pad(57:64) | h1(64:576) | h2(576:1088)]
#define KC1   18       // GEMM1 K = 576  (18 chunks of 32)
#define KC2   34       // GEMM2 K = 1088 (34 chunks of 32)
#define COL_W   3
#define COL_H1  64     // 16B-aligned LDS byte offset (128)
#define COL_H2  576    // 16B-aligned LDS byte offset (1152)
#define LDA   1104     // LDS row stride in bf16 elements (2208 B, 16B aligned)
#define NWG   16       // cooperating workgroups (each owns 32 hidden units)
#define NTHR  256      // 8 waves of 32

typedef __bf16 bf16_t;
typedef __attribute__((ext_vector_type(16))) __bf16 v16bf;
typedef __attribute__((ext_vector_type(8)))  __bf16 v8bf;
typedef __attribute__((ext_vector_type(8)))  float  v8f;

// pointer types for the async global->LDS builtin: (V4i addrspace(1)*, V4i addrspace(3)*, Ii, Ii)
typedef int v4i_t __attribute__((vector_size(16)));
typedef __attribute__((address_space(1))) v4i_t* gv4i_p;
typedef __attribute__((address_space(3))) v4i_t* lv4i_p;

#if defined(__has_builtin)
#  if __has_builtin(__builtin_amdgcn_global_load_async_to_lds_b128)
#    define HAVE_ASYNC_COPY 1
#  endif
#  if __has_builtin(__builtin_amdgcn_s_wait_asynccnt)
#    define HAVE_WAIT_ASYNC 1
#  endif
#endif

// ---------------- workspace layout (bytes) ----------------
constexpr size_t WS_SYNC = 0;                                   // 2 x u32 (cnt, gen)
constexpr size_t WS_B1   = 256;                                 // 2048 f32
constexpr size_t WS_B2   = WS_B1 + 2048 * 4;                    // 2048 f32
constexpr size_t WS_WK   = WS_B2 + 2048 * 4;                    // 30*512 bf16
constexpr size_t WS_WO   = WS_WK + 30 * 512 * 2;                // 121*512 bf16
constexpr size_t WS_WC1  = WS_WO + 121 * 512 * 2;               // swizzled LSTM1 weights
constexpr size_t WS_WC2  = WS_WC1 + (size_t)128 * KC1 * 32 * 16 * 2;
constexpr size_t WS_H1   = WS_WC2 + (size_t)128 * KC2 * 32 * 16 * 2;
constexpr size_t WS_H2   = WS_H1 + 64 * 512 * 2;                // h2 bf16

// dynamic LDS: A[64*LDA] bf16 | kappa[640] | alpha[640] | beta[640] | phi[4096] | outs[484]
constexpr unsigned SMEM_BYTES = 64 * LDA * 2 + 640 * 4 * 3 + 4096 * 4 + 4 * OUT_ * 4;

// ---------------- prep: biases + small weights to bf16 ----------------
__global__ void prep_misc(const float* __restrict__ b_ih1, const float* __restrict__ b_hh1,
                          const float* __restrict__ b_ih2, const float* __restrict__ b_hh2,
                          const float* __restrict__ W_k,   const float* __restrict__ W_o,
                          char* __restrict__ ws, float* __restrict__ out) {
  int i = blockIdx.x * 256 + threadIdx.x;
  if (i == 0) {
    unsigned* s = (unsigned*)(ws + WS_SYNC);
    s[0] = 0u; s[1] = 0u; out[0] = 0.f;
  }
  if (i < 2048) {
    ((float*)(ws + WS_B1))[i] = b_ih1[i] + b_hh1[i];
  } else if (i < 4096) {
    int j = i - 2048;
    ((float*)(ws + WS_B2))[j] = b_ih2[j] + b_hh2[j];
  } else if (i < 4096 + 30 * 512) {
    int j = i - 4096;
    ((bf16_t*)(ws + WS_WK))[j] = (bf16_t)W_k[j];
  } else if (i < 4096 + 30 * 512 + 121 * 512) {
    int j = i - 4096 - 30 * 512;
    ((bf16_t*)(ws + WS_WO))[j] = (bf16_t)W_o[j];
  }
}

// ---------------- prep: swizzle big weights into WMMA B-fragment layout ----------------
// dest element index = (((ntg*KC + kc)*32 + lane)*16 + e)
// lane -> N column of 16x16 tile; K position follows the 16-bit fragment layout:
//   half = lane>>4 ; v = e>>1 ; p = e&1
//   k = kc*32 + (v<4 ? half*8 + v*2 + p : 16 + half*8 + (v-4)*2 + p)
// Columns follow the unified activation layout; pad columns get zero weights so the
// padded-K WMMA chunks contribute exactly 0 regardless of A contents.
__global__ void swizzleW(bf16_t* __restrict__ dst, const float* __restrict__ Wih,
                         const float* __restrict__ Whh, int KC, int mode) {
  int idx = blockIdx.x * 256 + threadIdx.x;
  int total = 128 * KC * 32 * 16;
  if (idx >= total) return;
  int e    = idx & 15;
  int lane = (idx >> 4) & 31;
  int rest = idx >> 9;
  int kc   = rest % KC;
  int ntg  = rest / KC;
  int wg   = ntg >> 3;
  int ntl  = ntg & 7;
  int nl   = ntl * 16 + (lane & 15);      // WG-local column 0..127
  int gt   = nl >> 5;                     // gate i/f/g/o
  int jl   = nl & 31;                     // local hidden unit
  int row  = gt * 512 + (wg << 5) + jl;   // row of W (2048)
  int half = lane >> 4;
  int v    = e >> 1;
  int p    = e & 1;
  int koff = (v < 4) ? (half * 8 + v * 2 + p) : (16 + half * 8 + (v - 4) * 2 + p);
  int k    = kc * 32 + koff;
  float val = 0.f;
  if (mode == 1) {                        // LSTM1: Wih [2048,57] over [x|w], Whh over h1
    if (k < 57)                      val = Wih[row * 57 + k];
    else if (k >= COL_H1 && k < 576) val = Whh[row * 512 + (k - COL_H1)];
  } else {                                // LSTM2: Wih [2048,569] over [x|h1|w], Whh over h2
    if (k < 3)                        val = Wih[row * 569 + k];
    else if (k < 57)                  val = Wih[row * 569 + 515 + (k - 3)];      // w part
    else if (k >= COL_H1 && k < 576)  val = Wih[row * 569 + 3 + (k - COL_H1)];   // h1 part
    else if (k >= COL_H2 && k < 1088) val = Whh[row * 512 + (k - COL_H2)];       // h2 part
  }
  dst[idx] = (bf16_t)val;
}

// ---------------- grid barrier (agent scope, sense via generation counter) ----------------
__device__ __forceinline__ void gsync(unsigned* cnt, unsigned* gen) {
  __syncthreads();
  if (threadIdx.x == 0) {
    __threadfence();
    unsigned g = __hip_atomic_load(gen, __ATOMIC_RELAXED, __HIP_MEMORY_SCOPE_AGENT);
    unsigned a = __hip_atomic_fetch_add(cnt, 1u, __ATOMIC_ACQ_REL, __HIP_MEMORY_SCOPE_AGENT);
    if (a == NWG - 1) {
      __hip_atomic_store(cnt, 0u, __ATOMIC_RELAXED, __HIP_MEMORY_SCOPE_AGENT);
      __hip_atomic_fetch_add(gen, 1u, __ATOMIC_RELEASE, __HIP_MEMORY_SCOPE_AGENT);
    } else {
      while (__hip_atomic_load(gen, __ATOMIC_ACQUIRE, __HIP_MEMORY_SCOPE_AGENT) == g)
        __builtin_amdgcn_s_sleep(2);
    }
    __threadfence();
  }
  __syncthreads();
}

__device__ __forceinline__ float sigm(float x) { return 1.f / (1.f + expf(-x)); }

// ---- async global->LDS staging of a 64x512 bf16 matrix into strided A columns ----
// Each 16B chunk: row b = p>>6, chunk c = p&63. LDS dst is 16B aligned by layout.
__device__ __forceinline__ void stage_h(bf16_t* __restrict__ A, const bf16_t* __restrict__ src,
                                        int colbase, int tid) {
#if defined(HAVE_ASYNC_COPY)
  for (int p = tid; p < 4096; p += NTHR) {
    int b = p >> 6, c = p & 63;
    const char* g = (const char*)src + (size_t)p * 16;
    char*       l = (char*)A + (size_t)b * (LDA * 2) + colbase * 2 + c * 16;
    __builtin_amdgcn_global_load_async_to_lds_b128((gv4i_p)g, (lv4i_p)l, 0, 0);
  }
#  if defined(HAVE_WAIT_ASYNC)
  __builtin_amdgcn_s_wait_asynccnt(0);
#  else
  asm volatile("s_wait_asynccnt 0" ::: "memory");
#  endif
#else
  for (int p = tid; p < 64 * 512; p += NTHR)
    A[(p >> 9) * LDA + colbase + (p & 511)] = src[p];
#endif
}

// ---------------- fused WMMA GEMM + LSTM cell ----------------
// Wave wv: mt = wv&3 (16 batch rows), ng = wv>>2 (16 hidden units).
// acc[g] holds gate g for the SAME (batch,unit) lane mapping -> cell update is per-lane VALU.
// Single running weight pointer; the 4 gate streams live at compile-time byte offsets
// g*GSTRIDE (fits the 24-bit ioffset), so inner-loop address math is one 64-bit add.
template <int KC>
__device__ __forceinline__ void gemm_lstm(const bf16_t* __restrict__ A,
                                          const bf16_t* __restrict__ Wc,
                                          const float* __restrict__ bb,
                                          float* cs, bf16_t* __restrict__ hg,
                                          int wg, int mt, int ng, int lane,
                                          int lhalf, int l16, int jglob) {
  constexpr size_t GSTRIDE = (size_t)2 * KC * 32 * 16;   // elements between gate streams
  v8f acc[4] = {};
  const bf16_t* ap = A + (mt * 16 + l16) * LDA + lhalf * 8;
  const bf16_t* wp = Wc + ((size_t)((wg * 8 + ng) * KC) * 32 + lane) * 16;
#pragma unroll 2
  for (int kc = 0; kc < KC; ++kc) {
    v8bf lo = *(const v8bf*)(ap);
    v8bf hi = *(const v8bf*)(ap + 16);
    v16bf af = __builtin_shufflevector(lo, hi, 0, 1, 2, 3, 4, 5, 6, 7,
                                       8, 9, 10, 11, 12, 13, 14, 15);
    if (kc + 1 < KC) __builtin_prefetch(wp + 512, 0, 1);
    const v16bf b0 = *(const v16bf*)(wp);
    const v16bf b1 = *(const v16bf*)(wp + GSTRIDE);
    const v16bf b2 = *(const v16bf*)(wp + 2 * GSTRIDE);
    const v16bf b3 = *(const v16bf*)(wp + 3 * GSTRIDE);
    acc[0] = __builtin_amdgcn_wmma_f32_16x16x32_bf16(false, af, false, b0, (short)0, acc[0], false, false);
    acc[1] = __builtin_amdgcn_wmma_f32_16x16x32_bf16(false, af, false, b1, (short)0, acc[1], false, false);
    acc[2] = __builtin_amdgcn_wmma_f32_16x16x32_bf16(false, af, false, b2, (short)0, acc[2], false, false);
    acc[3] = __builtin_amdgcn_wmma_f32_16x16x32_bf16(false, af, false, b3, (short)0, acc[3], false, false);
    ap += 32;
    wp += 512;
  }
  const float bi = bb[jglob], bf = bb[512 + jglob], bg = bb[1024 + jglob], bo = bb[1536 + jglob];
#pragma unroll
  for (int r = 0; r < 8; ++r) {
    int b  = mt * 16 + lhalf * 8 + r;                  // batch row (C/D layout: lanes>=16 -> M+8)
    float iS = sigm(acc[0][r] + bi);
    float fS = sigm(acc[1][r] + bf);
    float gT = tanhf(acc[2][r] + bg);
    float oS = sigm(acc[3][r] + bo);
    cs[r] = fS * cs[r] + iS * gT;
    float h = oS * tanhf(cs[r]);
    hg[b * 512 + jglob] = (bf16_t)h;
  }
}

// ---------------- persistent recurrent kernel ----------------
__global__ __launch_bounds__(NTHR, 1)
void rnn_attn_kernel(const float* __restrict__ x, const float* __restrict__ y,
                     const float* __restrict__ c_vec,
                     const float* __restrict__ b_k, const float* __restrict__ b_o,
                     char* __restrict__ ws, float* __restrict__ out) {
  extern __shared__ char smem[];
  bf16_t* A   = (bf16_t*)smem;                         // [64][LDA] unified activations (bf16)
  float* kap  = (float*)(smem + 64 * LDA * 2);         // [64][10] persistent kappa
  float* alp  = kap + 640;
  float* bet  = alp + 640;
  float* phi  = bet + 640;                             // [64][64]
  float* outs = phi + 4096;                            // [4][121]

  unsigned* cnt = (unsigned*)(ws + WS_SYNC);
  unsigned* gen = cnt + 1;
  const float*  b1  = (const float*)(ws + WS_B1);
  const float*  b2  = (const float*)(ws + WS_B2);
  const bf16_t* Wk  = (const bf16_t*)(ws + WS_WK);
  const bf16_t* Wo  = (const bf16_t*)(ws + WS_WO);
  const bf16_t* Wc1 = (const bf16_t*)(ws + WS_WC1);
  const bf16_t* Wc2 = (const bf16_t*)(ws + WS_WC2);
  bf16_t* h1g = (bf16_t*)(ws + WS_H1);
  bf16_t* h2g = (bf16_t*)(ws + WS_H2);

  const int tid   = threadIdx.x;
  const int wg    = blockIdx.x;
  const int wv    = tid >> 5;
  const int lane  = tid & 31;
  const int lhalf = lane >> 4;
  const int l16   = lane & 15;
  const int mt    = wv & 3;
  const int ng    = wv >> 2;
  const int jglob = (wg << 5) + ng * 16 + l16;         // hidden unit owned by this lane

  float c1s[8], c2s[8];
#pragma unroll
  for (int r = 0; r < 8; ++r) { c1s[r] = 0.f; c2s[r] = 0.f; }
  float lacc = 0.f;

  // zero entire A (pad columns must be finite: padded weight chunks are zero) and kappa
  for (int p = tid; p < 64 * LDA; p += NTHR) A[p] = (bf16_t)0.0f;
  for (int p = tid; p < 640; p += NTHR) kap[p] = 0.f;
  __syncthreads();

  for (int t = 0; t < T_; ++t) {
    // ---- assemble x_t and h2(old) columns of A ----
    for (int p = tid; p < 64 * 3; p += NTHR) {
      int b = p / 3, c = p % 3;
      A[b * LDA + c] = (bf16_t)x[(b * T_ + t) * 3 + c];
    }
    if (t == 0) {
      for (int p = tid; p < 64 * 512; p += NTHR)
        A[(p >> 9) * LDA + COL_H2 + (p & 511)] = (bf16_t)0.0f;
    } else {
      stage_h(A, h2g, COL_H2, tid);    // async global->LDS if available
    }
    __syncthreads();

    // ---- LSTM1: gates = [x|w|0|h1] * Wc1^T  (K = 576) ----
    gemm_lstm<KC1>(A, Wc1, b1, c1s, h1g, wg, mt, ng, lane, lhalf, l16, jglob);
    gsync(cnt, gen);   // h1 complete device-wide

    // reload full new h1 into A[64:576)
    stage_h(A, h1g, COL_H1, tid);
    __syncthreads();

    // ---- attention window (computed redundantly per WG; kappa/w live in LDS) ----
    for (int p = tid; p < 640; p += NTHR) {
      int b = p / 10, k = p % 10;
      float ah = b_k[k], bh = b_k[10 + k], kh = b_k[20 + k];
      const bf16_t* hrow = &A[b * LDA + COL_H1];
      const bf16_t* w0 = &Wk[k * 512];
      const bf16_t* w1 = &Wk[(10 + k) * 512];
      const bf16_t* w2 = &Wk[(20 + k) * 512];
      for (int q = 0; q < 512; ++q) {
        float hv = (float)hrow[q];
        ah += hv * (float)w0[q];
        bh += hv * (float)w1[q];
        kh += hv * (float)w2[q];
      }
      alp[p] = expf(ah);
      bet[p] = expf(bh);
      kap[p] += expf(kh);
    }
    __syncthreads();
    for (int p = tid; p < 64 * 64; p += NTHR) {
      int b = p >> 6;
      float u = (float)(p & 63);
      float s = 0.f;
      for (int k = 0; k < 10; ++k) {
        float d = kap[b * 10 + k] - u;
        s += expf(-bet[b * 10 + k] * d * d) * alp[b * 10 + k];
      }
      phi[p] = s;
    }
    __syncthreads();
    for (int p = tid; p < 64 * 54; p += NTHR) {
      int b = p / 54, c = p % 54;
      float s = 0.f;
      const float* cv = &c_vec[(size_t)b * 64 * 54 + c];
      for (int u = 0; u < 64; ++u) s += phi[b * 64 + u] * cv[u * 54];
      A[b * LDA + COL_W + c] = (bf16_t)s;   // w(new) -> also reused as w(old) next step
    }
    __syncthreads();

    // ---- LSTM2: gates = [x|w|0|h1|h2] * Wc2^T  (K = 1088) ----
    gemm_lstm<KC2>(A, Wc2, b2, c2s, h2g, wg, mt, ng, lane, lhalf, l16, jglob);
    gsync(cnt, gen);   // h2 complete device-wide

    // ---- per-step output projection + MDN/Bernoulli loss (this WG owns 4 batch rows) ----
    for (int p = tid; p < 4 * OUT_; p += NTHR) {
      int q = p / OUT_, oi = p % OUT_;
      int b = wg * 4 + q;
      float s = b_o[oi];
      const bf16_t* hr = &h2g[b * 512];
      const bf16_t* wr = &Wo[oi * 512];
      for (int k2 = 0; k2 < 512; ++k2) s += (float)hr[k2] * (float)wr[k2];
      outs[q * OUT_ + oi] = s;
    }
    __syncthreads();
    if (tid < 4) {
      int b = wg * 4 + tid;
      const float* op = &outs[tid * OUT_];
      float y1 = y[(b * T_ + t) * 3 + 0];
      float y2 = y[(b * T_ + t) * 3 + 1];
      float sv = y[(b * T_ + t) * 3 + 2];
      float e  = 1.f / (1.f + expf(op[0]));            // sigmoid(-o0)
      float mx = -1e30f;
      for (int j = 0; j < 20; ++j) mx = fmaxf(mx, op[1 + j]);
      float den = 0.f;
      for (int j = 0; j < 20; ++j) den += expf(op[1 + j] - mx);
      float gsum = 0.f;
      for (int j = 0; j < 20; ++j) {
        float pi  = expf(op[1 + j] - mx) / den;
        float mu1 = op[21 + j], mu2 = op[41 + j];
        float s1  = expf(op[61 + j]);
        float s2  = expf(op[81 + j]);
        float rho = tanhf(op[101 + j]);
        float d1  = (y1 - mu1) / s1;
        float d2  = (y2 - mu2) / s2;
        float z   = d1 * d1 + d2 * d2 - 2.f * rho * d1 * d2;
        float omr = 1.f - rho * rho;
        float g   = expf(-z / (2.f * omr)) / (6.283185307179586f * s1 * s2 * sqrtf(omr));
        gsum += pi * g;
      }
      lacc += -logf(gsum + 1e-20f);
      lacc += -logf((e + 1e-20f) * sv + (1.f - e + 1e-20f) * (1.f - sv));
    }
    __syncthreads();
  }

  if (tid < 4) atomicAdd(out, lacc * (1.0f / (B_ * T_)));
}

// ---------------- host entry ----------------
extern "C" void kernel_launch(void* const* d_in, const int* in_sizes, int n_in,
                              void* d_out, int out_size, void* d_ws, size_t ws_size,
                              hipStream_t stream) {
  (void)in_sizes; (void)n_in; (void)out_size; (void)ws_size;
  const float* x     = (const float*)d_in[0];
  const float* y     = (const float*)d_in[1];
  const float* c_vec = (const float*)d_in[2];
  const float* W_ih1 = (const float*)d_in[3];
  const float* W_hh1 = (const float*)d_in[4];
  const float* b_ih1 = (const float*)d_in[5];
  const float* b_hh1 = (const float*)d_in[6];
  const float* W_ih2 = (const float*)d_in[7];
  const float* W_hh2 = (const float*)d_in[8];
  const float* b_ih2 = (const float*)d_in[9];
  const float* b_hh2 = (const float*)d_in[10];
  const float* W_k   = (const float*)d_in[11];
  const float* b_k   = (const float*)d_in[12];
  const float* W_o   = (const float*)d_in[13];
  const float* b_o   = (const float*)d_in[14];
  char*  ws  = (char*)d_ws;
  float* out = (float*)d_out;

  // stage 0: biases, small weights, sync counters, zero output
  prep_misc<<<318, 256, 0, stream>>>(b_ih1, b_hh1, b_ih2, b_hh2, W_k, W_o, ws, out);
  // stage 1: swizzle big weights into WMMA fragment layout (bf16)
  swizzleW<<<(128 * KC1 * 32 * 16) / 256, 256, 0, stream>>>(
      (bf16_t*)(ws + WS_WC1), W_ih1, W_hh1, KC1, 1);
  swizzleW<<<(128 * KC2 * 32 * 16) / 256, 256, 0, stream>>>(
      (bf16_t*)(ws + WS_WC2), W_ih2, W_hh2, KC2, 2);

  // stage 2: persistent cooperative recurrence (~167 KB dynamic LDS per WGP)
  (void)hipFuncSetAttribute((const void*)rnn_attn_kernel,
                            hipFuncAttributeMaxDynamicSharedMemorySize, SMEM_BYTES);
  rnn_attn_kernel<<<NWG, NTHR, SMEM_BYTES, stream>>>(x, y, c_vec, b_k, b_o, ws, out);
}
